// SingleEdit_12876311954014
// MI455X (gfx1250) — matrix operand order repaired and checked by reading, not compiled, not run on hardware
//
#include <hip/hip_runtime.h>
#include <cmath>

typedef _Float16 half_t;
typedef __attribute__((ext_vector_type(16))) _Float16 v16h;
typedef __attribute__((ext_vector_type(8)))  _Float16 v8h;
typedef __attribute__((ext_vector_type(4)))  _Float16 v4h;
typedef __attribute__((ext_vector_type(8)))  float    v8f;
typedef __attribute__((ext_vector_type(4)))  float    v4f;

#define MPN 256
#define MLP 512
#define BS_OUT 5
#define N_ATOMS 125001
#define N_BONDS 250000

// ---------------------------------------------------------------------------
// fp32 KxN -> f16 "B-fragment blocked": fragment f = kt*NT + nt,
// element e = (f*32 + lane)*16 + i holds W[kt*32+(lane>>4)*16+i][nt*16+(lane&15)]
// ---------------------------------------------------------------------------
__global__ void convert_blocked_f16(const float* __restrict__ w,
                                    half_t* __restrict__ dst, int K, int N) {
    const int NT = N >> 4;
    const int total = K * N;
    for (int e = blockIdx.x * blockDim.x + threadIdx.x; e < total;
         e += gridDim.x * blockDim.x) {
        const int i    = e & 15;
        const int lane = (e >> 4) & 31;
        const int frag = e >> 9;
        const int nt   = frag % NT;
        const int kt   = frag / NT;
        const int k = kt * 32 + (lane >> 4) * 16 + i;
        const int n = nt * 16 + (lane & 15);
        dst[e] = (half_t)w[(size_t)k * N + n];
    }
}

// fp32 KxNsrc -> f16 blocked with N padded to 16 (cols >= Nsrc are zero)
__global__ void convert_blocked_pad16_f16(const float* __restrict__ w,
                                          half_t* __restrict__ dst,
                                          int K, int Nsrc) {
    const int total = K * 16;
    for (int e = blockIdx.x * blockDim.x + threadIdx.x; e < total;
         e += gridDim.x * blockDim.x) {
        const int i    = e & 15;
        const int lane = (e >> 4) & 31;
        const int kt   = e >> 9;
        const int k = kt * 32 + (lane >> 4) * 16 + i;
        const int n = lane & 15;
        dst[e] = (n < Nsrc) ? (half_t)w[(size_t)k * Nsrc + n] : (half_t)0.0f;
    }
}

// A fragment (16x32 f16): lane = lhalf*16 + lrow, row M = lrow,
// halves 0..15 = K[kbase + lhalf*16 .. +15]
__device__ inline v16h load_a_frag(const half_t* __restrict__ base_row,
                                   int kbase, int lhalf) {
    const half_t* p = base_row + kbase + lhalf * 16;
    v8h lo = *(const v8h*)(p);
    v8h hi = *(const v8h*)(p + 8);
    return __builtin_shufflevector(lo, hi, 0, 1, 2, 3, 4, 5, 6, 7,
                                   8, 9, 10, 11, 12, 13, 14, 15);
}

// ---------------------------------------------------------------------------
// Bond MLP: 64 bonds per workgroup, 256 threads (8 waves)
// ---------------------------------------------------------------------------
__global__ __launch_bounds__(256)
void bond_mlp_kernel(const float* __restrict__ c_atom,
                     const int*   __restrict__ bond_index,
                     const half_t* __restrict__ bw1h,   // blocked f16 512x512
                     const float* __restrict__ bb1,
                     const half_t* __restrict__ bw2h,   // blocked f16 512x16 (5 valid)
                     const float* __restrict__ bb2,
                     float* __restrict__ out) {
    constexpr int TM  = 64;
    constexpr int K1  = 2 * MPN;       // 512
    constexpr int LDA = K1 + 8;        // 520 halves -> 1040B rows, 16B aligned
    constexpr int LDH = MLP + 8;       // 520
    constexpr int NT  = MLP / 16;      // 32

    __shared__ half_t sA[TM * LDA];    // 66,560 B
    __shared__ half_t sH[TM * LDH];    // 66,560 B
    __shared__ float  sOut[TM * 16];   //  4,096 B

    const int tid  = threadIdx.x;
    const int row0 = blockIdx.x * TM;

    // ---- Stage 1: gather endpoints, build [s+e, |s-e|]; init sOut with bias
    {
        const int r   = tid >> 2;
        const int c0  = (tid & 3) * 64;
        const int g   = row0 + r;
        half_t* arow = sA + r * LDA;
        if (g < N_BONDS) {
            const int i0 = bond_index[2 * g + 0];
            const int i1 = bond_index[2 * g + 1];
            const float* ps = c_atom + (size_t)i0 * MPN;
            const float* pe = c_atom + (size_t)i1 * MPN;
            #pragma unroll
            for (int c = 0; c < 64; c += 4) {
                v4f s = *(const v4f*)(ps + c0 + c);
                v4f e = *(const v4f*)(pe + c0 + c);
                v4h sv, dv;
                #pragma unroll
                for (int j = 0; j < 4; ++j) {
                    sv[j] = (half_t)(s[j] + e[j]);
                    dv[j] = (half_t)fabsf(s[j] - e[j]);
                }
                *(v4h*)(arow + c0 + c)       = sv;
                *(v4h*)(arow + MPN + c0 + c) = dv;
            }
        } else {
            #pragma unroll
            for (int c = 0; c < 64; c += 4) {
                *(v4h*)(arow + c0 + c)       = (v4h){};
                *(v4h*)(arow + MPN + c0 + c) = (v4h){};
            }
        }
        #pragma unroll
        for (int q = 0; q < 4; ++q) {
            const int idx = tid * 4 + q;
            const int c = idx & 15;
            sOut[idx] = (c < BS_OUT) ? bb2[c] : 0.f;
        }
    }
    __syncthreads();

    const int wave  = tid >> 5;
    const int lane  = tid & 31;
    const int lhalf = lane >> 4;
    const int lrow  = lane & 15;

    // ---- Stage 2: hidden = relu(A @ bw1 + bb1); two N-passes per wave ----
    #pragma unroll
    for (int pass = 0; pass < 2; ++pass) {
        v8f acc[2][4];
        #pragma unroll
        for (int p = 0; p < 2; ++p)
            #pragma unroll
            for (int m = 0; m < 4; ++m) acc[p][m] = (v8f){};

        const int ntb = wave * 4 + pass * 2;
        const half_t* bp0 = bw1h + ((size_t)(ntb + 0) * 32 + lane) * 16;
        const half_t* bp1 = bw1h + ((size_t)(ntb + 1) * 32 + lane) * 16;

        #pragma unroll 2
        for (int kt = 0; kt < K1 / 32; ++kt) {
            v16h a[4];
            #pragma unroll
            for (int mt = 0; mt < 4; ++mt)
                a[mt] = load_a_frag(sA + (mt * 16 + lrow) * LDA, kt * 32, lhalf);
            v16h bf0 = *(const v16h*)(bp0 + kt * (NT * 512));
            v16h bf1 = *(const v16h*)(bp1 + kt * (NT * 512));
            #pragma unroll
            for (int mt = 0; mt < 4; ++mt)
                acc[0][mt] = __builtin_amdgcn_wmma_f32_16x16x32_f16(
                    false, a[mt], false, bf0, (short)0, acc[0][mt], false, false);
            #pragma unroll
            for (int mt = 0; mt < 4; ++mt)
                acc[1][mt] = __builtin_amdgcn_wmma_f32_16x16x32_f16(
                    false, a[mt], false, bf1, (short)0, acc[1][mt], false, false);
        }
        #pragma unroll
        for (int ntp = 0; ntp < 2; ++ntp) {
            const int nt = ntb + ntp;
            const float bias = bb1[nt * 16 + lrow];
            #pragma unroll
            for (int mt = 0; mt < 4; ++mt) {
                #pragma unroll
                for (int v = 0; v < 8; ++v) {
                    float x = acc[ntp][mt][v] + bias;
                    x = x > 0.f ? x : 0.f;
                    sH[(mt * 16 + v + 8 * lhalf) * LDH + nt * 16 + lrow] =
                        (half_t)x;
                }
            }
        }
    }

    // ---- Stage 3 (GEMM2 via WMMA): each wave multiplies the H column slice
    // it just wrote (K-slice [wave*64, wave*64+64)) -- no barrier needed.
    {
        v8f acc2[4];
        #pragma unroll
        for (int m = 0; m < 4; ++m) acc2[m] = (v8f){};
        const half_t* b2 = bw2h + (((size_t)wave * 2) * 32 + lane) * 16;
        #pragma unroll
        for (int kk = 0; kk < 2; ++kk) {
            const int kbase = wave * 64 + kk * 32;
            v16h bf = *(const v16h*)(b2 + kk * 512);
            #pragma unroll
            for (int mt = 0; mt < 4; ++mt) {
                v16h af = load_a_frag(sH + (mt * 16 + lrow) * LDH, kbase, lhalf);
                acc2[mt] = __builtin_amdgcn_wmma_f32_16x16x32_f16(
                    false, af, false, bf, (short)0, acc2[mt], false, false);
            }
        }
        if (lrow < BS_OUT) {
            #pragma unroll
            for (int mt = 0; mt < 4; ++mt)
                #pragma unroll
                for (int v = 0; v < 8; ++v)
                    atomicAdd(&sOut[(mt * 16 + v + 8 * lhalf) * 16 + lrow],
                              acc2[mt][v]);
        }
    }
    __syncthreads();

    // ---- Stage 4: write logits ----
    for (int o = tid; o < TM * BS_OUT; o += 256) {
        const int r = o / BS_OUT, c = o % BS_OUT;
        const int g = row0 + r;
        if (g < N_BONDS) out[(size_t)g * BS_OUT + c] = sOut[r * 16 + c];
    }
}

// ---------------------------------------------------------------------------
// Unimol MLP: 64 atoms per workgroup, 256 threads (8 waves)
// ---------------------------------------------------------------------------
__global__ __launch_bounds__(256)
void unimol_mlp_kernel(const float* __restrict__ c_atom,
                       const half_t* __restrict__ uw1h,  // blocked f16 256x512
                       const float* __restrict__ ub1,
                       const half_t* __restrict__ uw2h,  // blocked f16 512x16 (1 valid)
                       const float* __restrict__ ub2,
                       float* __restrict__ out) {
    constexpr int TM  = 64;
    constexpr int K1  = MPN;           // 256
    constexpr int LDA = K1 + 8;        // 264
    constexpr int LDH = MLP + 8;       // 520
    constexpr int NT  = MLP / 16;      // 32

    __shared__ half_t sA[TM * LDA];    // 33,792 B
    __shared__ half_t sH[TM * LDH];    // 66,560 B
    __shared__ float  sOut[TM * 16];   //  4,096 B

    const int tid  = threadIdx.x;
    const int row0 = blockIdx.x * TM;

    // ---- Stage 1: load c_atom tile as f16; init sOut with bias ----
    {
        const int r  = tid >> 2;
        const int c0 = (tid & 3) * 64;
        const int g  = row0 + r;
        half_t* arow = sA + r * LDA;
        if (g < N_ATOMS) {
            const float* p = c_atom + (size_t)g * MPN;
            #pragma unroll
            for (int c = 0; c < 64; c += 4) {
                v4f x = *(const v4f*)(p + c0 + c);
                v4h h;
                #pragma unroll
                for (int j = 0; j < 4; ++j) h[j] = (half_t)x[j];
                *(v4h*)(arow + c0 + c) = h;
            }
        } else {
            #pragma unroll
            for (int c = 0; c < 64; c += 4) *(v4h*)(arow + c0 + c) = (v4h){};
        }
        #pragma unroll
        for (int q = 0; q < 4; ++q) {
            const int idx = tid * 4 + q;
            sOut[idx] = ((idx & 15) == 0) ? ub2[0] : 0.f;
        }
    }
    __syncthreads();

    const int wave  = tid >> 5;
    const int lane  = tid & 31;
    const int lhalf = lane >> 4;
    const int lrow  = lane & 15;

    #pragma unroll
    for (int pass = 0; pass < 2; ++pass) {
        v8f acc[2][4];
        #pragma unroll
        for (int p = 0; p < 2; ++p)
            #pragma unroll
            for (int m = 0; m < 4; ++m) acc[p][m] = (v8f){};

        const int ntb = wave * 4 + pass * 2;
        const half_t* bp0 = uw1h + ((size_t)(ntb + 0) * 32 + lane) * 16;
        const half_t* bp1 = uw1h + ((size_t)(ntb + 1) * 32 + lane) * 16;

        #pragma unroll 2
        for (int kt = 0; kt < K1 / 32; ++kt) {
            v16h a[4];
            #pragma unroll
            for (int mt = 0; mt < 4; ++mt)
                a[mt] = load_a_frag(sA + (mt * 16 + lrow) * LDA, kt * 32, lhalf);
            v16h bf0 = *(const v16h*)(bp0 + kt * (NT * 512));
            v16h bf1 = *(const v16h*)(bp1 + kt * (NT * 512));
            #pragma unroll
            for (int mt = 0; mt < 4; ++mt)
                acc[0][mt] = __builtin_amdgcn_wmma_f32_16x16x32_f16(
                    false, a[mt], false, bf0, (short)0, acc[0][mt], false, false);
            #pragma unroll
            for (int mt = 0; mt < 4; ++mt)
                acc[1][mt] = __builtin_amdgcn_wmma_f32_16x16x32_f16(
                    false, a[mt], false, bf1, (short)0, acc[1][mt], false, false);
        }
        #pragma unroll
        for (int ntp = 0; ntp < 2; ++ntp) {
            const int nt = ntb + ntp;
            const float bias = ub1[nt * 16 + lrow];
            #pragma unroll
            for (int mt = 0; mt < 4; ++mt) {
                #pragma unroll
                for (int v = 0; v < 8; ++v) {
                    float x = acc[ntp][mt][v] + bias;
                    x = x > 0.f ? x : 0.f;
                    sH[(mt * 16 + v + 8 * lhalf) * LDH + nt * 16 + lrow] =
                        (half_t)x;
                }
            }
        }
    }

    // ---- Stage 3 (GEMM2 via WMMA): own K-slice, partials into sOut ----
    {
        v8f acc2[4];
        #pragma unroll
        for (int m = 0; m < 4; ++m) acc2[m] = (v8f){};
        const half_t* b2 = uw2h + (((size_t)wave * 2) * 32 + lane) * 16;
        #pragma unroll
        for (int kk = 0; kk < 2; ++kk) {
            const int kbase = wave * 64 + kk * 32;
            v16h bf = *(const v16h*)(b2 + kk * 512);
            #pragma unroll
            for (int mt = 0; mt < 4; ++mt) {
                v16h af = load_a_frag(sH + (mt * 16 + lrow) * LDH, kbase, lhalf);
                acc2[mt] = __builtin_amdgcn_wmma_f32_16x16x32_f16(
                    false, af, false, bf, (short)0, acc2[mt], false, false);
            }
        }
        if (lrow == 0) {
            #pragma unroll
            for (int mt = 0; mt < 4; ++mt)
                #pragma unroll
                for (int v = 0; v < 8; ++v)
                    atomicAdd(&sOut[(mt * 16 + v + 8 * lhalf) * 16],
                              acc2[mt][v]);
        }
    }
    __syncthreads();

    if (tid < TM) {
        const int g = row0 + tid;
        if (g < N_ATOMS) out[g] = sOut[tid * 16];
    }
}

extern "C" void kernel_launch(void* const* d_in, const int* in_sizes, int n_in,
                              void* d_out, int out_size, void* d_ws, size_t ws_size,
                              hipStream_t stream) {
    const float* c_atom     = (const float*)d_in[0];
    const int*   bond_index = (const int*)d_in[1];
    const float* bw1 = (const float*)d_in[2];
    const float* bb1 = (const float*)d_in[3];
    const float* bw2 = (const float*)d_in[4];
    const float* bb2 = (const float*)d_in[5];
    const float* uw1 = (const float*)d_in[6];
    const float* ub1 = (const float*)d_in[7];
    const float* uw2 = (const float*)d_in[8];
    const float* ub2 = (const float*)d_in[9];
    float* out = (float*)d_out;

    half_t* bw1h = (half_t*)d_ws;                  // 512*512 f16 = 512 KB
    half_t* uw1h = bw1h + (size_t)512 * 512;       // 256*512 f16 = 256 KB
    half_t* bw2h = uw1h + (size_t)256 * 512;       // 512*16  f16 =  16 KB
    half_t* uw2h = bw2h + (size_t)512 * 16;        // 512*16  f16 =  16 KB

    convert_blocked_f16<<<1024, 256, 0, stream>>>(bw1, bw1h, 2 * MPN, MLP);
    convert_blocked_f16<<<512,  256, 0, stream>>>(uw1, uw1h, MPN, MLP);
    convert_blocked_pad16_f16<<<32, 256, 0, stream>>>(bw2, bw2h, MLP, BS_OUT);
    convert_blocked_pad16_f16<<<32, 256, 0, stream>>>(uw2, uw2h, MLP, 1);

    bond_mlp_kernel<<<(N_BONDS + 63) / 64, 256, 0, stream>>>(
        c_atom, bond_index, bw1h, bb1, bw2h, bb2, out);
    unimol_mlp_kernel<<<(N_ATOMS + 63) / 64, 256, 0, stream>>>(
        c_atom, uw1h, ub1, uw2h, ub2, out + (size_t)N_BONDS * BS_OUT);
}
